// Dispersion_loss_10788957848170
// MI455X (gfx1250) — matrix operand order repaired and checked by reading, not compiled, not run on hardware
//
#include <hip/hip_runtime.h>

// ---------------- types ----------------
typedef __attribute__((ext_vector_type(16))) __bf16          v16bf;
typedef __attribute__((ext_vector_type(8)))  float           v8f;
typedef __attribute__((ext_vector_type(8)))  unsigned short  ushort8;
typedef __attribute__((ext_vector_type(4)))  float           float4v;

#define N_CLS   8192
#define DIM     2048
#define BT      128                         // block tile edge (M and N)
#define KT      32                          // K chunk (bf16 WMMA depth)
#define NCH     (DIM / KT)                  // 64 K chunks
#define LDSW    40                          // padded LDS row stride (elems) -> 80B, conflict-free
#define NTILE   (N_CLS / BT)                // 64
#define NPAIRS  (NTILE * (NTILE + 1) / 2)   // 2080 upper-triangular tiles

// ---------------- bf16 helpers (no header-struct dependence) ----------------
__device__ __forceinline__ unsigned short f32_to_bf16_rne(float f) {
  unsigned u = __float_as_uint(f);
  u += 0x7FFFu + ((u >> 16) & 1u);   // round-to-nearest-even
  return (unsigned short)(u >> 16);
}
__device__ __forceinline__ float bf16_to_f32(unsigned short h) {
  return __uint_as_float(((unsigned)h) << 16);
}

union Frag { ushort8 h[2]; v16bf v; };

// triangular tile decode (uniform per block): 0 <= ti <= tj < NTILE
__device__ __forceinline__ void decode_tile(unsigned t, unsigned& ti, unsigned& tj) {
  const unsigned u = (unsigned)(NPAIRS - 1) - t;
  unsigned b = (unsigned)((sqrtf(8.f * (float)u + 1.f) - 1.f) * 0.5f);
  while ((b + 1) * (b + 2) / 2 <= u) ++b;
  while (b * (b + 1) / 2 > u) --b;
  ti = (NTILE - 1) - b;
  tj = (NTILE - 1) - (u - b * (b + 1) / 2);
}

// ---------------- kernel 1a: row norms only (fallback path) ----------------
__global__ __launch_bounds__(256)
void rownorm_kernel(const float* __restrict__ x,
                    float* __restrict__ inv, float* __restrict__ sq) {
  __shared__ float red[256];
  const int row = blockIdx.x;
  const float* xr = x + (size_t)row * DIM;

  float s = 0.f;
  for (int c = threadIdx.x; c < DIM; c += 256) { float v = xr[c]; s += v * v; }
  red[threadIdx.x] = s; __syncthreads();
  for (int off = 128; off > 0; off >>= 1) {
    if (threadIdx.x < off) red[threadIdx.x] += red[threadIdx.x + off];
    __syncthreads();
  }
  const float invn = 1.0f / fmaxf(sqrtf(red[0]), 1e-12f);
  __syncthreads();

  float s2 = 0.f;
  for (int c = threadIdx.x; c < DIM; c += 256) { float v = xr[c] * invn; s2 += v * v; }
  red[threadIdx.x] = s2; __syncthreads();
  for (int off = 128; off > 0; off >>= 1) {
    if (threadIdx.x < off) red[threadIdx.x] += red[threadIdx.x + off];
    __syncthreads();
  }
  if (threadIdx.x == 0) { inv[row] = invn; sq[row] = red[0]; }
}

// ---------------- kernel 1b: row norms + bf16 hi/lo split (fast path) ----------------
__global__ __launch_bounds__(256)
void rownorm_split_kernel(const float* __restrict__ x,
                          float* __restrict__ inv, float* __restrict__ sq,
                          unsigned short* __restrict__ xhi,
                          unsigned short* __restrict__ xlo) {
  __shared__ float red[256];
  const int row = blockIdx.x;
  const float* xr = x + (size_t)row * DIM;

  float s = 0.f;
  for (int c = threadIdx.x; c < DIM; c += 256) { float v = xr[c]; s += v * v; }
  red[threadIdx.x] = s; __syncthreads();
  for (int off = 128; off > 0; off >>= 1) {
    if (threadIdx.x < off) red[threadIdx.x] += red[threadIdx.x + off];
    __syncthreads();
  }
  const float invn = 1.0f / fmaxf(sqrtf(red[0]), 1e-12f);
  __syncthreads();

  float s2 = 0.f;
  for (int c = threadIdx.x; c < DIM; c += 256) {
    const float v = xr[c] * invn;
    const unsigned short h = f32_to_bf16_rne(v);
    xhi[(size_t)row * DIM + c] = h;
    xlo[(size_t)row * DIM + c] = f32_to_bf16_rne(v - bf16_to_f32(h));
    s2 += v * v;
  }
  red[threadIdx.x] = s2; __syncthreads();
  for (int off = 128; off > 0; off >>= 1) {
    if (threadIdx.x < off) red[threadIdx.x] += red[threadIdx.x + off];
    __syncthreads();
  }
  if (threadIdx.x == 0) { inv[row] = invn; sq[row] = red[0]; }
}

// ---------------- shared epilogue: exp + deterministic reductions ----------------
__device__ __forceinline__ void tile_epilogue(
    const v8f acc[4][2], const float* __restrict__ sq, float* __restrict__ partials,
    float* red, int tid, int rowA0, int rowB0, int waveM, int waveN,
    int lr, int lh, unsigned t, unsigned ti, unsigned tj) {
  float lsum = 0.f;
  #pragma unroll
  for (int mi = 0; mi < 4; ++mi) {
    const int gmb = rowA0 + waveM + mi * 16 + lh * 8;   // C/D: M = vgpr + 8*(lane/16)
    #pragma unroll
    for (int ni = 0; ni < 2; ++ni) {
      const int gn = rowB0 + waveN + ni * 16 + lr;      // N = lane%16
      const float sqn = sq[gn];
      #pragma unroll
      for (int r = 0; r < 8; ++r) {
        const float g  = acc[mi][ni][r];
        const float d2 = fmaxf(sq[gmb + r] + sqn - 2.0f * g, 0.0f);
        lsum += __expf(-d2);
      }
    }
  }
  red[tid] = lsum; __syncthreads();
  for (int off = 128; off > 0; off >>= 1) {
    if (tid < off) red[tid] += red[tid + off];
    __syncthreads();
  }
  if (tid == 0) partials[t] = (ti == tj) ? red[0] : 2.0f * red[0];
}

// issue one K-chunk of async bf16 tile loads: wave w -> array (w&3), row-half (w>>2)
// arrays: 0=Ahi 1=Alo 2=Bhi 3=Blo ; each wave issues 8x b128 (512B each) = 4KB
__device__ __forceinline__ void issue_chunk_async(
    const unsigned short* __restrict__ xhi, const unsigned short* __restrict__ xlo,
    int rowA0, int rowB0, int kk, unsigned ldsBase, int wave, int lane) {
  const int arr  = wave & 3;
  const int half = wave >> 2;
  const unsigned short* src = (arr & 1) ? xlo : xhi;
  const int row0 = (arr & 2) ? rowB0 : rowA0;
  const int r    = half * 64 + (lane >> 2);     // this lane's row (8-row step per instr)
  const int quad = (lane & 3) * 16;             // 16B slice within 64B row
  const char* g = (const char*)(src + (size_t)(row0 + r) * DIM + kk) + quad;
  unsigned l = ldsBase + (unsigned)arr * (BT * LDSW * 2)
                       + (unsigned)r * (LDSW * 2) + (unsigned)quad;
  #pragma unroll
  for (int j = 0; j < 8; ++j) {
    asm volatile("global_load_async_to_lds_b128 %0, %1, off"
                 :: "v"(l), "v"(g) : "memory");
    g += (size_t)8 * DIM * 2;   // +8 global rows
    l += 8 * (LDSW * 2);        // +8 LDS rows
  }
}

// ---------------- kernel 2a: fast path — async LDS pipeline, no conversions ----------------
__global__ __launch_bounds__(256)
void gram_exp_async_kernel(const unsigned short* __restrict__ xhi,
                           const unsigned short* __restrict__ xlo,
                           const float* __restrict__ sq,
                           float* __restrict__ partials) {
  __shared__ unsigned short sbuf[2][4][BT][LDSW];   // [buf][Ahi,Alo,Bhi,Blo]
  __shared__ float red[256];

  unsigned ti, tj;
  const unsigned t = blockIdx.x;
  decode_tile(t, ti, tj);
  const int rowA0 = (int)ti * BT, rowB0 = (int)tj * BT;

  const int tid  = threadIdx.x;
  const int wave = tid >> 5, lane = tid & 31;
  const int waveM = (wave >> 2) * 64;
  const int waveN = (wave & 3) * 32;
  const int lr = lane & 15, lh = lane >> 4;
  const unsigned ldsBase0 = (unsigned)(size_t)&sbuf[0][0][0][0];
  const unsigned ldsBase1 = (unsigned)(size_t)&sbuf[1][0][0][0];

  v8f acc[4][2];
  {
    const v8f z = {0.f,0.f,0.f,0.f,0.f,0.f,0.f,0.f};
    #pragma unroll
    for (int mi = 0; mi < 4; ++mi)
      #pragma unroll
      for (int ni = 0; ni < 2; ++ni) acc[mi][ni] = z;
  }

  issue_chunk_async(xhi, xlo, rowA0, rowB0, 0, ldsBase0, wave, lane);

  for (int i = 0; i < NCH; ++i) {
    const int cur = i & 1;
    asm volatile("s_wait_asynccnt 0x0" ::: "memory");  // our chunk-i loads landed
    __syncthreads();                                    // everyone's landed; prev compute done
    if (i + 1 < NCH)
      issue_chunk_async(xhi, xlo, rowA0, rowB0, (i + 1) * KT,
                        cur ? ldsBase0 : ldsBase1, wave, lane);

    const unsigned short (*sAhi)[LDSW] = sbuf[cur][0];
    const unsigned short (*sAlo)[LDSW] = sbuf[cur][1];
    const unsigned short (*sBhi)[LDSW] = sbuf[cur][2];
    const unsigned short (*sBlo)[LDSW] = sbuf[cur][3];

    // B fragments (32x16 bf16: lane<16 -> K0..7,K16..23 ; lane>=16 -> K8..15,K24..31)
    Frag bhi[2], blo[2];
    #pragma unroll
    for (int ni = 0; ni < 2; ++ni) {
      const int br = waveN + ni * 16 + lr;
      const int kb = lh * 8;
      bhi[ni].h[0] = *(const ushort8*)&sBhi[br][kb];
      bhi[ni].h[1] = *(const ushort8*)&sBhi[br][16 + kb];
      blo[ni].h[0] = *(const ushort8*)&sBlo[br][kb];
      blo[ni].h[1] = *(const ushort8*)&sBlo[br][16 + kb];
    }
    // A fragments (16x32 bf16: lane<16 -> K0..7,K16..23 ; lane>=16 -> K8..15,K24..31)
    #pragma unroll
    for (int mi = 0; mi < 4; ++mi) {
      const int ar = waveM + mi * 16 + lr;
      const int ka = lh * 8;
      Frag ahi, alo;
      ahi.h[0] = *(const ushort8*)&sAhi[ar][ka];
      ahi.h[1] = *(const ushort8*)&sAhi[ar][16 + ka];
      alo.h[0] = *(const ushort8*)&sAlo[ar][ka];
      alo.h[1] = *(const ushort8*)&sAlo[ar][16 + ka];
      #pragma unroll
      for (int ni = 0; ni < 2; ++ni) {
        // bf16x3: hi*hi + hi*lo + lo*hi  (fp32 accumulate)
        acc[mi][ni] = __builtin_amdgcn_wmma_f32_16x16x32_bf16(
            false, ahi.v, false, bhi[ni].v, (short)0, acc[mi][ni], false, false);
        acc[mi][ni] = __builtin_amdgcn_wmma_f32_16x16x32_bf16(
            false, ahi.v, false, blo[ni].v, (short)0, acc[mi][ni], false, false);
        acc[mi][ni] = __builtin_amdgcn_wmma_f32_16x16x32_bf16(
            false, alo.v, false, bhi[ni].v, (short)0, acc[mi][ni], false, false);
      }
    }
  }

  tile_epilogue(acc, sq, partials, red, tid, rowA0, rowB0, waveM, waveN, lr, lh, t, ti, tj);
}

// ---------------- kernel 2b: fallback — convert-on-the-fly (round-1 design) ----------------
__global__ __launch_bounds__(256)
void gram_exp_fly_kernel(const float* __restrict__ x,
                         const float* __restrict__ inv,
                         const float* __restrict__ sq,
                         float* __restrict__ partials) {
  __shared__ unsigned short sAhi[BT][LDSW];
  __shared__ unsigned short sAlo[BT][LDSW];
  __shared__ unsigned short sBhi[BT][LDSW];
  __shared__ unsigned short sBlo[BT][LDSW];
  __shared__ float red[256];

  unsigned ti, tj;
  const unsigned t = blockIdx.x;
  decode_tile(t, ti, tj);
  const int rowA0 = (int)ti * BT, rowB0 = (int)tj * BT;

  const int tid  = threadIdx.x;
  const int wave = tid >> 5, lane = tid & 31;
  const int waveM = (wave >> 2) * 64;
  const int waveN = (wave & 3) * 32;
  const int lr = lane & 15, lh = lane >> 4;

  v8f acc[4][2];
  {
    const v8f z = {0.f,0.f,0.f,0.f,0.f,0.f,0.f,0.f};
    #pragma unroll
    for (int mi = 0; mi < 4; ++mi)
      #pragma unroll
      for (int ni = 0; ni < 2; ++ni) acc[mi][ni] = z;
  }

  for (int kk = 0; kk < DIM; kk += KT) {
    __syncthreads();
    #pragma unroll
    for (int i = 0; i < 4; ++i) {
      const int q = tid + 256 * i;
      const int r = q >> 3;
      const int c = (q & 7) * 4;
      {
        const int gr = rowA0 + r;
        const float iv = inv[gr];
        float4v v = *(const float4v*)(x + (size_t)gr * DIM + kk + c);
        #pragma unroll
        for (int e = 0; e < 4; ++e) {
          const float xv = v[e] * iv;
          const unsigned short h = f32_to_bf16_rne(xv);
          sAhi[r][c + e] = h;
          sAlo[r][c + e] = f32_to_bf16_rne(xv - bf16_to_f32(h));
        }
      }
      {
        const int gr = rowB0 + r;
        const float iv = inv[gr];
        float4v v = *(const float4v*)(x + (size_t)gr * DIM + kk + c);
        #pragma unroll
        for (int e = 0; e < 4; ++e) {
          const float xv = v[e] * iv;
          const unsigned short h = f32_to_bf16_rne(xv);
          sBhi[r][c + e] = h;
          sBlo[r][c + e] = f32_to_bf16_rne(xv - bf16_to_f32(h));
        }
      }
    }
    if (kk + KT < DIM) {
      const int pr = tid >> 1;
      const int pc = (tid & 1) * 16;
      __builtin_prefetch(x + (size_t)(rowA0 + pr) * DIM + kk + KT + pc, 0, 1);
      __builtin_prefetch(x + (size_t)(rowB0 + pr) * DIM + kk + KT + pc, 0, 1);
    }
    __syncthreads();

    Frag bhi[2], blo[2];
    #pragma unroll
    for (int ni = 0; ni < 2; ++ni) {
      const int br = waveN + ni * 16 + lr;
      const int kb = lh * 8;
      bhi[ni].h[0] = *(const ushort8*)&sBhi[br][kb];
      bhi[ni].h[1] = *(const ushort8*)&sBhi[br][16 + kb];
      blo[ni].h[0] = *(const ushort8*)&sBlo[br][kb];
      blo[ni].h[1] = *(const ushort8*)&sBlo[br][16 + kb];
    }
    #pragma unroll
    for (int mi = 0; mi < 4; ++mi) {
      const int ar = waveM + mi * 16 + lr;
      const int ka = lh * 8;
      Frag ahi, alo;
      ahi.h[0] = *(const ushort8*)&sAhi[ar][ka];
      ahi.h[1] = *(const ushort8*)&sAhi[ar][16 + ka];
      alo.h[0] = *(const ushort8*)&sAlo[ar][ka];
      alo.h[1] = *(const ushort8*)&sAlo[ar][16 + ka];
      #pragma unroll
      for (int ni = 0; ni < 2; ++ni) {
        acc[mi][ni] = __builtin_amdgcn_wmma_f32_16x16x32_bf16(
            false, ahi.v, false, bhi[ni].v, (short)0, acc[mi][ni], false, false);
        acc[mi][ni] = __builtin_amdgcn_wmma_f32_16x16x32_bf16(
            false, ahi.v, false, blo[ni].v, (short)0, acc[mi][ni], false, false);
        acc[mi][ni] = __builtin_amdgcn_wmma_f32_16x16x32_bf16(
            false, alo.v, false, bhi[ni].v, (short)0, acc[mi][ni], false, false);
      }
    }
  }

  tile_epilogue(acc, sq, partials, red, tid, rowA0, rowB0, waveM, waveN, lr, lh, t, ti, tj);
}

// ---------------- kernel 3: deterministic final reduction ----------------
__global__ __launch_bounds__(256)
void final_reduce_kernel(const float* __restrict__ partials, float* __restrict__ out) {
  __shared__ float red[256];
  float s = 0.f;
  for (int i = threadIdx.x; i < NPAIRS; i += 256) s += partials[i];
  red[threadIdx.x] = s; __syncthreads();
  for (int off = 128; off > 0; off >>= 1) {
    if (threadIdx.x < off) red[threadIdx.x] += red[threadIdx.x + off];
    __syncthreads();
  }
  if (threadIdx.x == 0)
    out[0] = red[0] / ((float)N_CLS * (float)(N_CLS - 1));
}

// ---------------- launcher ----------------
extern "C" void kernel_launch(void* const* d_in, const int* in_sizes, int n_in,
                              void* d_out, int out_size, void* d_ws, size_t ws_size,
                              hipStream_t stream) {
  (void)in_sizes; (void)n_in; (void)out_size;
  const float* x = (const float*)d_in[0];

  // small scratch (always fits)
  float* inv      = (float*)d_ws;          // 8192 floats
  float* sq       = inv + N_CLS;           // 8192 floats
  float* partials = sq + N_CLS;            // 2080 floats

  // fast path needs 2x bf16 split arrays (64 MB) after a 128KB header
  const size_t hdr    = 128 * 1024;
  const size_t splitb = (size_t)N_CLS * DIM * 2;        // 32 MB each
  const size_t need   = hdr + 2 * splitb;

  if (ws_size >= need) {
    unsigned short* xhi = (unsigned short*)((char*)d_ws + hdr);
    unsigned short* xlo = (unsigned short*)((char*)d_ws + hdr + splitb);
    rownorm_split_kernel<<<N_CLS, 256, 0, stream>>>(x, inv, sq, xhi, xlo);
    gram_exp_async_kernel<<<NPAIRS, 256, 0, stream>>>(xhi, xlo, sq, partials);
  } else {
    rownorm_kernel<<<N_CLS, 256, 0, stream>>>(x, inv, sq);
    gram_exp_fly_kernel<<<NPAIRS, 256, 0, stream>>>(x, inv, sq, partials);
  }
  final_reduce_kernel<<<1, 256, 0, stream>>>(partials, (float*)d_out);
}
